// SimpleMultiHeadAttention_850403525340
// MI455X (gfx1250) — matrix operand order repaired and checked by reading, not compiled, not run on hardware
//
#include <hip/hip_runtime.h>
#include <hip/hip_bf16.h>

// ---------------------------------------------------------------------------
// SimpleMultiHeadAttention for MI455X (gfx1250), wave32 + WMMA f16->f32.
// Pipeline:  f32->f16 convert pass -> Q/K/V = x @ W^T (async-LDS GEMM)
//            -> flash attention -> out = AO @ Wo^T + bo (f32, d_out)
// ---------------------------------------------------------------------------

#define DM 2048      // d_model
#define TT 2048      // sequence length
#define NB 2         // batch
#define NH 16        // heads
#define DH 128       // head dim

typedef __attribute__((ext_vector_type(16))) _Float16 v16h;
typedef __attribute__((ext_vector_type(8)))  _Float16 hk8;
typedef __attribute__((ext_vector_type(4)))  _Float16 hk4;
typedef __attribute__((ext_vector_type(8)))  float    v8f;

// D = A(16x32 f16) x B(32x16 f16) + C(16x16 f32)
__device__ __forceinline__ v8f wmma16(v16h a, v16h b, v8f c) {
  return __builtin_amdgcn_wmma_f32_16x16x32_f16(
      /*neg_a=*/false, a, /*neg_b=*/false, b,
      /*c_mod=*/(short)0, c, /*reuse_a=*/false, /*reuse_b=*/false);
}

// Build a 16-half fragment from two 8-half (16B) chunks (LDS or global).
__device__ __forceinline__ v16h ldfrag(const _Float16* p0, const _Float16* p1) {
  const hk8 lo = *(const hk8*)p0;
  const hk8 hi = *(const hk8*)p1;
  v16h r;
#pragma unroll
  for (int i = 0; i < 8; ++i) { r[i] = lo[i]; r[i + 8] = hi[i]; }
  return r;
}

// Column gather: 16 halves with an element stride (B operand from row-major V).
__device__ __forceinline__ v16h ldfrag_col(const _Float16* p, int stride) {
  v16h r;
#pragma unroll
  for (int i = 0; i < 16; ++i) r[i] = p[i * stride];
  return r;
}

// --- CDNA5 async global->LDS (tracked by ASYNCcnt) -------------------------
// LDS operand: wave-relative byte offset = low 32 bits of a generic pointer
// into __shared__ (ISA 10.2: LDS aperture keeps the offset in addr[31:0]).
__device__ __forceinline__ unsigned lds_off(const void* p) {
  return (unsigned)(unsigned long long)p;
}
__device__ __forceinline__ void async_b128(unsigned lds, const void* g) {
  asm volatile("global_load_async_to_lds_b128 %0, %1, off"
               :: "v"(lds), "v"((unsigned long long)g) : "memory");
}
__device__ __forceinline__ void wait_async0() {
  asm volatile("s_wait_asynccnt 0x0" ::: "memory");
}

// ---------------------------------------------------------------------------
// f32 -> f16 elementwise convert (one-shot pass; everything then sits in L2)
// ---------------------------------------------------------------------------
__global__ __launch_bounds__(256) void conv_f32_f16_kernel(
    const float* __restrict__ s, _Float16* __restrict__ d, int n4) {
  const int i = blockIdx.x * 256 + threadIdx.x;
  if (i < n4) {
    const float4 v = ((const float4*)s)[i];
    hk4 t;
    t[0] = (_Float16)v.x; t[1] = (_Float16)v.y;
    t[2] = (_Float16)v.z; t[3] = (_Float16)v.w;
    ((hk4*)d)[i] = t;
  }
}

// ---------------------------------------------------------------------------
// Y[M,N] = X[M,K] @ W[N,K]^T (+ bias), all-f16 operands, M = NB*TT, N = K = DM.
// Block: 256 thr (8 waves), tile 128x128, K-step 32, double-buffered LDS fed
// by async global->LDS DMA overlapped with compute. Wave (wm,wn) owns a 32x64
// strip (2 m-tiles x 4 n-tiles = 8 WMMAs per K-step from 6 fragment loads).
// ---------------------------------------------------------------------------
template <typename TOUT, bool BIAS>
__global__ __launch_bounds__(256) void gemm_nt_kernel(
    const _Float16* __restrict__ X, const _Float16* __restrict__ W,
    const float* __restrict__ bias, TOUT* __restrict__ Y) {
  __shared__ _Float16 sX[2][128 * 40];   // 40-half stride: 80B rows (16B aligned)
  __shared__ _Float16 sW[2][128 * 40];

  const int tid  = threadIdx.x;
  const int lane = tid & 31;
  const int wave = tid >> 5;
  const int wm = wave >> 1, wn = wave & 1;     // 4 x 2 wave grid
  const int lh = lane >> 4, ln = lane & 15;
  const long bm = (long)blockIdx.y * 128;
  const long bn = (long)blockIdx.x * 128;

  // async staging: each thread DMAs 16 halves (2 x b128) of one row per matrix
  const int arow = tid >> 1;          // 0..127
  const int ac   = (tid & 1) * 16;    // 0 or 16 halves
  const _Float16* gXr = X + (bm + arow) * (long)DM + ac;
  const _Float16* gWr = W + (bn + arow) * (long)DM + ac;

  // issue tile 0 -> buffer 0
#pragma unroll
  for (int j = 0; j < 2; ++j) {
    async_b128(lds_off(&sX[0][arow * 40 + ac + j * 8]), gXr + j * 8);
    async_b128(lds_off(&sW[0][arow * 40 + ac + j * 8]), gWr + j * 8);
  }

  v8f acc[2][4] = {};

  for (int kt = 0; kt < DM / 32; ++kt) {
    const int bsel = kt & 1;
    wait_async0();                    // own DMA for tile kt has landed
    __syncthreads();                  // all waves landed + prev reads done
    if (kt + 1 < DM / 32) {           // DMA tile kt+1 into the other buffer
      const int nb = bsel ^ 1;
      const long ko = (long)(kt + 1) * 32;
#pragma unroll
      for (int j = 0; j < 2; ++j) {
        async_b128(lds_off(&sX[nb][arow * 40 + ac + j * 8]), gXr + ko + j * 8);
        async_b128(lds_off(&sW[nb][arow * 40 + ac + j * 8]), gWr + ko + j * 8);
      }
    }

    v16h a[2], bb[4];
#pragma unroll
    for (int mt = 0; mt < 2; ++mt) {   // A: M = ln, K = {lh*8+0..7, 16+lh*8+0..7}
      const _Float16* pa = &sX[bsel][(wm * 32 + mt * 16 + ln) * 40 + lh * 8];
      a[mt] = ldfrag(pa, pa + 16);
    }
#pragma unroll
    for (int nt = 0; nt < 4; ++nt) {   // B: N = ln, K = {lh*8+0..7, 16+lh*8+0..7}
      const _Float16* pb = &sW[bsel][(wn * 64 + nt * 16 + ln) * 40 + lh * 8];
      bb[nt] = ldfrag(pb, pb + 16);
    }
#pragma unroll
    for (int mt = 0; mt < 2; ++mt)
#pragma unroll
      for (int nt = 0; nt < 4; ++nt)
        acc[mt][nt] = wmma16(a[mt], bb[nt], acc[mt][nt]);
  }

  // C/D layout: M = v + 8*lh, N = ln
#pragma unroll
  for (int mt = 0; mt < 2; ++mt)
#pragma unroll
    for (int nt = 0; nt < 4; ++nt) {
      const long col = bn + wn * 64 + nt * 16 + ln;
      const float bv = BIAS ? bias[col] : 0.0f;
#pragma unroll
      for (int v = 0; v < 8; ++v) {
        const long row = bm + wm * 32 + mt * 16 + 8 * lh + v;
        Y[row * (long)DM + col] = (TOUT)(acc[mt][nt][v] + bv);
      }
    }
}

// ---------------------------------------------------------------------------
// Flash attention: 1 block = (b, h, 64 q rows); 128 thr = 4 waves,
// each wave owns 16 q rows (private online-softmax state in VGPRs).
// Q A-fragments load straight from global (no LDS); K/V staged to LDS with
// register prefetch of the next 64-key tile.
// ---------------------------------------------------------------------------
__global__ __launch_bounds__(128) void attn_kernel(
    const _Float16* __restrict__ Qp, const _Float16* __restrict__ Kp,
    const _Float16* __restrict__ Vp, _Float16* __restrict__ AO) {
  __shared__ _Float16 sK[64 * 136];    // 64 key rows x 128 d (272B rows)
  __shared__ _Float16 sV[64 * 136];    // 64 key rows x 128 d (row-major)
  __shared__ _Float16 sP[4 * 16 * 72]; // per-wave P: 16 q x 64 key (144B rows)

  const int tid  = threadIdx.x;
  const int lane = tid & 31;
  const int wave = tid >> 5;
  const int lh = lane >> 4, ln = lane & 15;

  const int bid = blockIdx.x;
  const int qt = bid & 31;             // T/64 = 32 q-tiles
  const int h  = (bid >> 5) & 15;
  const int b  = bid >> 9;

  const long qrow0 = (long)b * TT + qt * 64;
  const long krow0 = (long)b * TT;
  const int  col0  = h * DH;

  // Q A-fragments for the whole loop, direct from global (16B-aligned chunks)
  v16h qa[4];
  {
    const _Float16* qrow = Qp + (qrow0 + wave * 16 + ln) * (long)DM + col0;
#pragma unroll
    for (int ks = 0; ks < 4; ++ks) {
      const _Float16* pq = qrow + ks * 32 + lh * 8;
      qa[ks] = ldfrag(pq, pq + 16);
    }
  }

  // staging coords: each thread owns 64 contiguous halves of one key row
  const int str = tid >> 1;
  const int stc = (tid & 1) * 64;

  hk8 kreg[8], vreg[8];
  {
    const long grow = krow0 + str;
    const _Float16* srcK = Kp + grow * (long)DM + col0 + stc;
    const _Float16* srcV = Vp + grow * (long)DM + col0 + stc;
#pragma unroll
    for (int j = 0; j < 8; ++j) {
      kreg[j] = *(const hk8*)(srcK + j * 8);
      vreg[j] = *(const hk8*)(srcV + j * 8);
    }
  }

  float m[8], l[8];
  v8f o[8] = {};
#pragma unroll
  for (int v = 0; v < 8; ++v) { m[v] = -1e30f; l[v] = 0.0f; }

  const float SCALE = 0.08838834764831845f;   // 1/sqrt(128)
  const float L2E   = 1.4426950408889634f;

  for (int kt = 0; kt < TT / 64; ++kt) {
    __syncthreads();                    // all waves done reading prev K/V tile
    {
      _Float16* dstK = &sK[str * 136 + stc];
      _Float16* dstV = &sV[str * 136 + stc];
#pragma unroll
      for (int j = 0; j < 8; ++j) {
        *(hk8*)(dstK + j * 8) = kreg[j];
        *(hk8*)(dstV + j * 8) = vreg[j];
      }
    }
    if (kt + 1 < TT / 64) {             // prefetch next key tile into regs
      const long grow = krow0 + (kt + 1) * 64 + str;
      const _Float16* srcK = Kp + grow * (long)DM + col0 + stc;
      const _Float16* srcV = Vp + grow * (long)DM + col0 + stc;
#pragma unroll
      for (int j = 0; j < 8; ++j) {
        kreg[j] = *(const hk8*)(srcK + j * 8);
        vreg[j] = *(const hk8*)(srcV + j * 8);
      }
    }
    __syncthreads();

    // S = Q K^T : 4 n-tiles x 4 k-steps = 16 WMMAs
    v8f s[4] = {};
#pragma unroll
    for (int t = 0; t < 4; ++t) {
#pragma unroll
      for (int ks = 0; ks < 4; ++ks) {
        const _Float16* pb = &sK[(t * 16 + ln) * 136 + ks * 32 + lh * 8];
        const v16h bk = ldfrag(pb, pb + 16);
        s[t] = wmma16(qa[ks], bk, s[t]);
      }
    }

    // online softmax (row M = v + 8*lh lives across the 16 lanes of a half)
    float al[8];
#pragma unroll
    for (int v = 0; v < 8; ++v) {
#pragma unroll
      for (int t = 0; t < 4; ++t) s[t][v] *= SCALE;
      float rm = fmaxf(fmaxf(s[0][v], s[1][v]), fmaxf(s[2][v], s[3][v]));
#pragma unroll
      for (int off = 1; off < 16; off <<= 1)
        rm = fmaxf(rm, __shfl_xor(rm, off, 32));
      const float mn = fmaxf(m[v], rm);
      al[v] = exp2f((m[v] - mn) * L2E);
      m[v] = mn;
      float ps = 0.0f;
#pragma unroll
      for (int t = 0; t < 4; ++t) {
        const float p = exp2f((s[t][v] - mn) * L2E);
        s[t][v] = p;
        ps += p;
      }
#pragma unroll
      for (int off = 1; off < 16; off <<= 1)
        ps += __shfl_xor(ps, off, 32);
      l[v] = l[v] * al[v] + ps;
    }
#pragma unroll
    for (int dt = 0; dt < 8; ++dt)
#pragma unroll
      for (int v = 0; v < 8; ++v) o[dt][v] *= al[v];

    // P: C-layout -> per-wave LDS -> A-layout (same-wave DS ops are in-order)
    _Float16* pw = &sP[wave * 16 * 72];
#pragma unroll
    for (int t = 0; t < 4; ++t)
#pragma unroll
      for (int v = 0; v < 8; ++v)
        pw[(8 * lh + v) * 72 + t * 16 + ln] = (_Float16)s[t][v];

    // O += P @ V : B frags gathered down columns of row-major sV
#pragma unroll
    for (int ks = 0; ks < 2; ++ks) {
      const _Float16* pa = &pw[ln * 72 + ks * 32 + lh * 16];
      const v16h af = ldfrag(pa, pa + 8);
#pragma unroll
      for (int dt = 0; dt < 8; ++dt) {
        const v16h bf =
            ldfrag_col(&sV[(ks * 32 + lh * 16) * 136 + dt * 16 + ln], 136);
        o[dt] = wmma16(af, bf, o[dt]);
      }
    }
  }

  // finalize: O /= l, write f16 attention output [B*T, DM]
#pragma unroll
  for (int v = 0; v < 8; ++v) {
    const float inv = 1.0f / l[v];
    const long arow = qrow0 + wave * 16 + 8 * lh + v;
#pragma unroll
    for (int dt = 0; dt < 8; ++dt)
      AO[arow * (long)DM + col0 + dt * 16 + ln] = (_Float16)(o[dt][v] * inv);
  }
}

// ---------------------------------------------------------------------------
extern "C" void kernel_launch(void* const* d_in, const int* in_sizes, int n_in,
                              void* d_out, int out_size, void* d_ws, size_t ws_size,
                              hipStream_t stream) {
  (void)in_sizes; (void)n_in; (void)out_size; (void)ws_size;
  const float* q  = (const float*)d_in[0];
  const float* k  = (const float*)d_in[1];
  const float* v  = (const float*)d_in[2];
  const float* Wq = (const float*)d_in[3];
  const float* Wk = (const float*)d_in[4];
  const float* Wv = (const float*)d_in[5];
  const float* Wo = (const float*)d_in[6];
  const float* bo = (const float*)d_in[7];
  float* out = (float*)d_out;

  // ws layout (f16): x16{q,k,v} 8M each, W16{q,k,v,o} 4M each,
  //                  Qp/Kp/Vp/AO 8M each  -> 72M halves = 144 MB
  const size_t MAT  = (size_t)NB * TT * DM;   // 8M elements
  const size_t WMAT = (size_t)DM * DM;        // 4M elements
  _Float16* p    = (_Float16*)d_ws;
  _Float16* X16q = p; p += MAT;
  _Float16* X16k = p; p += MAT;
  _Float16* X16v = p; p += MAT;
  _Float16* W16q = p; p += WMAT;
  _Float16* W16k = p; p += WMAT;
  _Float16* W16v = p; p += WMAT;
  _Float16* W16o = p; p += WMAT;
  _Float16* Qp   = p; p += MAT;
  _Float16* Kp   = p; p += MAT;
  _Float16* Vp   = p; p += MAT;
  _Float16* AO   = p;

  // one-shot f32 -> f16 conversion pass
  const int n4x = (int)(MAT / 4), n4w = (int)(WMAT / 4);
  conv_f32_f16_kernel<<<n4x / 256, 256, 0, stream>>>(q,  X16q, n4x);
  conv_f32_f16_kernel<<<n4x / 256, 256, 0, stream>>>(k,  X16k, n4x);
  conv_f32_f16_kernel<<<n4x / 256, 256, 0, stream>>>(v,  X16v, n4x);
  conv_f32_f16_kernel<<<n4w / 256, 256, 0, stream>>>(Wq, W16q, n4w);
  conv_f32_f16_kernel<<<n4w / 256, 256, 0, stream>>>(Wk, W16k, n4w);
  conv_f32_f16_kernel<<<n4w / 256, 256, 0, stream>>>(Wv, W16v, n4w);
  conv_f32_f16_kernel<<<n4w / 256, 256, 0, stream>>>(Wo, W16o, n4w);

  const dim3 gg(DM / 128, (NB * TT) / 128);   // (16, 32)
  gemm_nt_kernel<_Float16, false><<<gg, 256, 0, stream>>>(X16q, W16q, nullptr, Qp);
  gemm_nt_kernel<_Float16, false><<<gg, 256, 0, stream>>>(X16k, W16k, nullptr, Kp);
  gemm_nt_kernel<_Float16, false><<<gg, 256, 0, stream>>>(X16v, W16v, nullptr, Vp);

  attn_kernel<<<NB * NH * (TT / 64), 128, 0, stream>>>(Qp, Kp, Vp, AO);

  gemm_nt_kernel<float, true><<<gg, 256, 0, stream>>>(AO, W16o, bo, out);
}
